// MultiheadFlashLinearEnsemble1_12695923327087
// MI455X (gfx1250) — matrix operand order repaired and checked by reading, not compiled, not run on hardware
//
#include <hip/hip_runtime.h>

// ---------------- problem constants ----------------
constexpr int BB = 2, TT = 2048, EE = 2048, HH = 16, KVH = 4, MM = 4, HD = 32;
constexpr int MROWS = BB * TT;               // 4096
constexpr int NKV   = EE / (HH / KVH);       // 512 (k/v projection width)

typedef __attribute__((ext_vector_type(16))) __bf16 v16bf;
typedef __attribute__((ext_vector_type(8)))  float  v8f;
typedef __attribute__((ext_vector_type(16))) unsigned short v16us;
typedef __attribute__((ext_vector_type(8)))  unsigned short v8us;
typedef __attribute__((ext_vector_type(4)))  unsigned short v4us;
typedef __attribute__((ext_vector_type(4)))  unsigned int   v4u;
typedef __attribute__((ext_vector_type(8)))  int            v8i;
typedef __attribute__((ext_vector_type(4)))  int            v4i;

union U16x16 { v16us v; v8us h[2]; unsigned short s[16]; };
union F32x8  { v8f v; float f[8]; };

#if __has_builtin(__builtin_amdgcn_tensor_load_to_lds) && \
    __has_builtin(__builtin_amdgcn_s_wait_tensorcnt)
#define HAVE_TDM 1
#else
#define HAVE_TDM 0
#endif

__device__ __forceinline__ unsigned short f2bf(float x) {
  unsigned int u = __builtin_bit_cast(unsigned int, x);
  u += 0x7FFFu + ((u >> 16) & 1u);           // round-to-nearest-even
  return (unsigned short)(u >> 16);
}

__device__ __forceinline__ v8f wmma_bf16(v16bf a, v16bf b, v8f c) {
  return __builtin_amdgcn_wmma_f32_16x16x32_bf16(false, a, false, b, (short)0, c,
                                                 false, false);
}

#if HAVE_TDM
// LDS aperture: flat addr low 32 bits == LDS byte offset (ISA 10.2)
__device__ __forceinline__ unsigned int lds_addr32(const void* p) {
  return (unsigned int)(unsigned long long)p;
}

// 2D TDM tile load: tile_d0 elements (bf16) per row, tile_d1 rows,
// global row stride = stride0 elements. dw0flags carries data_size/pad codes.
__device__ __forceinline__ void tdm_load_2d(unsigned int ldsoff, const void* gp,
                                            unsigned int tile_d0, unsigned int tile_d1,
                                            unsigned long long stride0,
                                            unsigned int dw0flags) {
  unsigned long long ga = (unsigned long long)gp;
  v4u g0;
  g0[0] = 1u;                                             // count=1, user descriptor
  g0[1] = ldsoff;                                         // LDS byte address
  g0[2] = (unsigned int)(ga & 0xFFFFFFFFu);               // global addr lo
  g0[3] = (unsigned int)((ga >> 32) & 0x01FFFFFFu) | (2u << 30);  // addr hi | type=2
  v8i g1;
  g1[0] = (int)dw0flags;                                  // data_size | pad codes
  g1[1] = (int)((0xFFFFFu & 0xFFFFu) << 16);              // tensor_dim0 lo16 (big)
  g1[2] = (int)(((0xFFFFFu >> 16) & 0xFFFFu) | ((0xFFFFFu & 0xFFFFu) << 16));
  g1[3] = (int)(((0xFFFFFu >> 16) & 0xFFFFu) | (tile_d0 << 16));  // dim1 hi | tile_d0
  g1[4] = (int)(tile_d1 & 0xFFFFu);                       // tile_d1, tile_d2=0
  g1[5] = (int)(unsigned int)(stride0 & 0xFFFFFFFFull);   // dim0 stride lo32
  g1[6] = (int)(unsigned int)((stride0 >> 32) & 0xFFFFull); // stride hi16, dim1 stride=0
  g1[7] = 0;
  v4i z4 = {0, 0, 0, 0};
  v8i z8 = {0, 0, 0, 0, 0, 0, 0, 0};
  __builtin_amdgcn_tensor_load_to_lds(g0, g1, z4, z4, z8, 0);
}
// data_size=2B; pad every 16 DWORDs by 4 DWORDs -> 40-half LDS row stride
#define TDM_FLAGS_LDP40  ((1u << 16) | (1u << 20) | (3u << 22) | (3u << 25))
// data_size=2B; pad every 64 DWORDs by 2 DWORDs -> 132-half LDS row stride
#define TDM_FLAGS_LDP132 ((1u << 16) | (1u << 20) | (5u << 22) | (1u << 25))
#endif

// ---------------- elementwise convert ----------------
__global__ void f32_to_bf16_k(const float* __restrict__ src,
                              unsigned short* __restrict__ dst, size_t n) {
  size_t i = (size_t)blockIdx.x * blockDim.x + threadIdx.x;
  size_t stride = (size_t)gridDim.x * blockDim.x;
  for (; i < n; i += stride) dst[i] = f2bf(src[i]);
}

// ---------------- RoPE (interleaved) + bf16 convert ----------------
__global__ void rope_to_bf16(const float* __restrict__ src,
                             const float* __restrict__ cosb,
                             const float* __restrict__ sinb,
                             unsigned short* __restrict__ dst,
                             int nheads, size_t npairs) {
  size_t i = (size_t)blockIdx.x * blockDim.x + threadIdx.x;
  if (i >= npairs) return;
  int d2 = (int)(i & 15);
  int t  = (int)(((i >> 4) / (size_t)nheads) % (size_t)TT);
  float x1 = src[2 * i], x2 = src[2 * i + 1];
  float c = cosb[t * (HD / 2) + d2], s = sinb[t * (HD / 2) + d2];
  dst[2 * i]     = f2bf(x1 * c - x2 * s);
  dst[2 * i + 1] = f2bf(x1 * s + x2 * c);
}

// ---------------- WMMA GEMM: C[m][n] = sum_k A[m][k] * W[n][k] ----------------
#define BMT 128
#define BNT 64
#define BKT 32
#define LDP 40   // padded LDS row stride (halves)

__global__ __launch_bounds__(256) void gemm_bf16_wmma(
    const unsigned short* __restrict__ A,
    const unsigned short* __restrict__ W,
    float* __restrict__ C, int N, int K) {
  __shared__ unsigned short As[2][BMT][LDP];
  __shared__ unsigned short Bs[2][BNT][LDP];
  const int tid  = threadIdx.x;
  const int lane = tid & 31;
  const int l16  = lane & 15;
  const int lh   = lane >> 4;
  const int wave = tid >> 5;
  const int wm   = wave >> 1;
  const int wn   = wave & 1;
  const int bm   = blockIdx.y * BMT;
  const int bn   = blockIdx.x * BNT;

  v8f acc[2][2] = {};

#if HAVE_TDM
  // prologue: TDM-stream tile 0 into buffer 0 (wave0: A, wave1: B)
  if (wave == 0)
    tdm_load_2d(lds_addr32(&As[0][0][0]), A + (size_t)bm * K, BKT, BMT,
                (unsigned long long)K, TDM_FLAGS_LDP40);
  if (wave == 1)
    tdm_load_2d(lds_addr32(&Bs[0][0][0]), W + (size_t)bn * K, BKT, BNT,
                (unsigned long long)K, TDM_FLAGS_LDP40);
  if (wave < 2) __builtin_amdgcn_s_wait_tensorcnt(0);
  __syncthreads();

  int cur = 0;
  for (int k0 = 0; k0 < K; k0 += BKT) {
    const int nxt = cur ^ 1;
    if (k0 + BKT < K) {             // overlap next tile DMA with this tile's WMMAs
      if (wave == 0)
        tdm_load_2d(lds_addr32(&As[nxt][0][0]), A + (size_t)bm * K + k0 + BKT,
                    BKT, BMT, (unsigned long long)K, TDM_FLAGS_LDP40);
      if (wave == 1)
        tdm_load_2d(lds_addr32(&Bs[nxt][0][0]), W + (size_t)bn * K + k0 + BKT,
                    BKT, BNT, (unsigned long long)K, TDM_FLAGS_LDP40);
    }
#else
  int cur = 0;
  for (int k0 = 0; k0 < K; k0 += BKT) {
    const int nxt = cur;
    // manual staging fallback
#pragma unroll
    for (int c = 0; c < 4; ++c) {
      int chunk = tid + c * 256;
      int row = chunk >> 3, col4 = (chunk & 7) << 2;
      *(v4us*)&As[0][row][col4] =
          *(const v4us*)(A + (size_t)(bm + row) * K + k0 + col4);
    }
#pragma unroll
    for (int c = 0; c < 2; ++c) {
      int chunk = tid + c * 256;
      int row = chunk >> 3, col4 = (chunk & 7) << 2;
      *(v4us*)&Bs[0][row][col4] =
          *(const v4us*)(W + (size_t)(bn + row) * K + k0 + col4);
    }
    __syncthreads();
#endif

    v16bf af[2], bfr[2];
#pragma unroll
    for (int r = 0; r < 2; ++r) {
      U16x16 t;
      const unsigned short* p = &As[cur][wm * 32 + r * 16 + l16][lh ? 8 : 0];
      t.h[0] = *(const v8us*)p;
      t.h[1] = *(const v8us*)(p + 16);
      af[r] = __builtin_bit_cast(v16bf, t.v);
    }
#pragma unroll
    for (int c = 0; c < 2; ++c) {
      U16x16 t;
      const unsigned short* p = &Bs[cur][wn * 32 + c * 16 + l16][lh * 8];
      t.h[0] = *(const v8us*)p;
      t.h[1] = *(const v8us*)(p + 16);
      bfr[c] = __builtin_bit_cast(v16bf, t.v);
    }
#pragma unroll
    for (int r = 0; r < 2; ++r)
#pragma unroll
      for (int c = 0; c < 2; ++c)
        acc[r][c] = wmma_bf16(af[r], bfr[c], acc[r][c]);

#if HAVE_TDM
    if (wave < 2) __builtin_amdgcn_s_wait_tensorcnt(0);
#endif
    __syncthreads();
    cur = nxt;
  }

#pragma unroll
  for (int r = 0; r < 2; ++r)
#pragma unroll
    for (int c = 0; c < 2; ++c) {
      F32x8 t; t.v = acc[r][c];
#pragma unroll
      for (int e = 0; e < 8; ++e) {
        int row = bm + wm * 32 + r * 16 + e + lh * 8;
        int col = bn + wn * 32 + c * 16 + l16;
        C[(size_t)row * N + col] = t.f[e];
      }
    }
}

// ---------------- fused 4-member causal flash attention ----------------
__global__ __launch_bounds__(128) void flash_ensemble(
    const unsigned short* __restrict__ Q,
    const unsigned short* __restrict__ Kt,
    const unsigned short* __restrict__ V,
    const float* __restrict__ raw_mw,
    const float* __restrict__ wscale,
    float* __restrict__ comb_out) {
  __shared__ unsigned short vtile[2][32][132];
  __shared__ unsigned short pbuf[4][16][40];
  __shared__ float comb[16][128];

  const int tid  = threadIdx.x;
  const int lane = tid & 31;
  const int l16  = lane & 15;
  const int lh   = lane >> 4;
  const int mi   = tid >> 5;                   // ensemble member = wave id
  const int q0   = blockIdx.x * 16;
  const int bh   = blockIdx.y;
  const int b    = bh >> 4;
  const int h    = bh & 15;
  const int kh   = h >> 2;

  for (int i = tid; i < 16 * 128; i += 128) ((float*)comb)[i] = 0.0f;

  const unsigned short* vbase = V + (size_t)(b * TT) * (size_t)NKV + (size_t)kh * (MM * HD);

  // Q A-fragment, register-resident for the whole pass
  v16bf qa;
  {
    const unsigned short* p = Q + (size_t)(b * TT + q0 + l16) * (size_t)EE
                                + (size_t)h * (MM * HD) + (size_t)mi * HD + (lh ? 8 : 0);
    U16x16 t;
    t.h[0] = *(const v8us*)p;
    t.h[1] = *(const v8us*)(p + 16);
    qa = __builtin_bit_cast(v16bf, t.v);
  }

  const v8f vzero = {0.f, 0.f, 0.f, 0.f, 0.f, 0.f, 0.f, 0.f};
  float mrun[8], lrun[8];
  F32x8 o[8];
#pragma unroll
  for (int e = 0; e < 8; ++e) { mrun[e] = -1e30f; lrun[e] = 0.0f; }
#pragma unroll
  for (int c = 0; c < 8; ++c) o[c].v = vzero;

  const float scale = 0.1767766952966369f;     // HD^-0.5
  const int nkb = (q0 + 16 + 31) >> 5;
  int cur = 0;

#if HAVE_TDM
  if (mi == 0) {
    tdm_load_2d(lds_addr32(&vtile[0][0][0]), vbase, MM * HD, 32,
                (unsigned long long)NKV, TDM_FLAGS_LDP132);
    __builtin_amdgcn_s_wait_tensorcnt(0);
  }
  __syncthreads();
#endif

  for (int kb = 0; kb < nkb; ++kb) {
    const int kbase = kb * 32;
    const int nxt = cur ^ 1;

#if HAVE_TDM
    if (kb + 1 < nkb && mi == 0)
      tdm_load_2d(lds_addr32(&vtile[nxt][0][0]), vbase + (size_t)(kbase + 32) * NKV,
                  MM * HD, 32, (unsigned long long)NKV, TDM_FLAGS_LDP132);
#else
    __syncthreads();
    {
      const unsigned short* vs = vbase + (size_t)kbase * NKV;
#pragma unroll
      for (int c = 0; c < 8; ++c) {
        int chunk = tid + c * 128;
        int row = chunk >> 5, col4 = (chunk & 31) << 2;
        *(v4us*)&vtile[0][row][col4] = *(const v4us*)(vs + (size_t)row * NKV + col4);
      }
    }
    __syncthreads();
#endif

    // K B-fragments straight from global (d contiguous); prefetch next block
    v16bf kf[2];
#pragma unroll
    for (int j = 0; j < 2; ++j) {
      const unsigned short* p = Kt + (size_t)(b * TT + kbase + j * 16 + l16) * (size_t)NKV
                                   + (size_t)kh * (MM * HD) + (size_t)mi * HD + lh * 8;
      if (kb + 1 < nkb) __builtin_prefetch(p + (size_t)32 * NKV, 0, 0);
      U16x16 t;
      t.h[0] = *(const v8us*)p;
      t.h[1] = *(const v8us*)(p + 16);
      kf[j] = __builtin_bit_cast(v16bf, t.v);
    }

    // S = Q K^T
    F32x8 s[2];
    s[0].v = wmma_bf16(qa, kf[0], vzero);
    s[1].v = wmma_bf16(qa, kf[1], vzero);

    // scale + causal mask + block row-max
    float mb[8];
#pragma unroll
    for (int e = 0; e < 8; ++e) {
      int rowg = q0 + e + lh * 8;
      float v0 = s[0].f[e] * scale;
      float v1 = s[1].f[e] * scale;
      if (kbase + l16 > rowg)      v0 = -1e30f;
      if (kbase + 16 + l16 > rowg) v1 = -1e30f;
      s[0].f[e] = v0; s[1].f[e] = v1;
      mb[e] = fmaxf(v0, v1);
    }
#pragma unroll
    for (int d = 1; d < 16; d <<= 1)
#pragma unroll
      for (int e = 0; e < 8; ++e)
        mb[e] = fmaxf(mb[e], __shfl_xor(mb[e], d, 32));

    // online softmax
    float alpha[8], rs[8];
#pragma unroll
    for (int e = 0; e < 8; ++e) {
      float mn = fmaxf(mrun[e], mb[e]);
      alpha[e] = __expf(mrun[e] - mn);
      mrun[e] = mn;
      float p0 = __expf(s[0].f[e] - mn);
      float p1 = __expf(s[1].f[e] - mn);
      s[0].f[e] = p0; s[1].f[e] = p1;
      rs[e] = p0 + p1;
    }
#pragma unroll
    for (int d = 1; d < 16; d <<= 1)
#pragma unroll
      for (int e = 0; e < 8; ++e)
        rs[e] += __shfl_xor(rs[e], d, 32);
#pragma unroll
    for (int e = 0; e < 8; ++e) lrun[e] = lrun[e] * alpha[e] + rs[e];
#pragma unroll
    for (int c = 0; c < 8; ++c)
#pragma unroll
      for (int e = 0; e < 8; ++e) o[c].f[e] *= alpha[e];

    // P: C-layout -> bf16 A-layout via per-wave LDS transpose
#pragma unroll
    for (int j = 0; j < 2; ++j)
#pragma unroll
      for (int e = 0; e < 8; ++e)
        pbuf[mi][e + lh * 8][j * 16 + l16] = f2bf(s[j].f[e]);

    v16bf pa;
    {
      const unsigned short* p = &pbuf[mi][l16][lh ? 8 : 0];
      U16x16 t;
      t.h[0] = *(const v8us*)p;
      t.h[1] = *(const v8us*)(p + 16);
      pa = __builtin_bit_cast(v16bf, t.v);
    }

    // O += P * V
#pragma unroll
    for (int c = 0; c < 8; ++c) {
      U16x16 t;
      int col = c * 16 + l16;
#pragma unroll
      for (int r = 0; r < 16; ++r) t.s[r] = vtile[cur][lh * 8 + (r & 7) + (r >> 3) * 16][col];
      o[c].v = wmma_bf16(pa, __builtin_bit_cast(v16bf, t.v), o[c].v);
    }

#if HAVE_TDM
    if (mi == 0) __builtin_amdgcn_s_wait_tensorcnt(0);
    __syncthreads();
    cur = nxt;
#else
    (void)nxt;
#endif
  }

  // ensemble combine: comb += map_w[mi] * O / l
  const float mw = tanhf(raw_mw[mi]) * wscale[0];
#pragma unroll
  for (int c = 0; c < 8; ++c)
#pragma unroll
    for (int e = 0; e < 8; ++e) {
      int row = e + lh * 8;
      int col = c * 16 + l16;
      atomicAdd(&comb[row][col], o[c].f[e] / lrun[e] * mw);
    }
  __syncthreads();

  float* dst = comb_out + (size_t)(b * TT + q0) * (size_t)EE + (size_t)h * (MM * HD);
  for (int i = tid; i < 16 * 128; i += 128) {
    int row = i >> 7, col = i & 127;
    dst[(size_t)row * EE + col] = comb[row][col];
  }
}

// ---------------- RMS norm over last dim 128, write bf16 ----------------
__global__ __launch_bounds__(128) void rmsnorm_bf16(
    const float* __restrict__ comb, const float* __restrict__ gamma,
    unsigned short* __restrict__ normed) {
  __shared__ float ssum[4];
  const size_t row = blockIdx.x;
  const int e = threadIdx.x;
  float v = comb[row * 128 + e];
  float sq = v * v;
#pragma unroll
  for (int d = 16; d >= 1; d >>= 1) sq += __shfl_xor(sq, d, 32);
  if ((e & 31) == 0) ssum[e >> 5] = sq;
  __syncthreads();
  float tot = ssum[0] + ssum[1] + ssum[2] + ssum[3];
  float r = rsqrtf(tot * (1.0f / 128.0f) + 1e-5f);
  normed[row * 128 + e] = f2bf(v * r * gamma[e]);
}

// ---------------- launch ----------------
extern "C" void kernel_launch(void* const* d_in, const int* in_sizes, int n_in,
                              void* d_out, int out_size, void* d_ws, size_t ws_size,
                              hipStream_t stream) {
  (void)in_sizes; (void)n_in; (void)out_size; (void)ws_size;
  const float* x     = (const float*)d_in[0];
  const float* cosb  = (const float*)d_in[1];
  const float* sinb  = (const float*)d_in[2];
  const float* q_w   = (const float*)d_in[3];
  const float* k_w   = (const float*)d_in[4];
  const float* v_w   = (const float*)d_in[5];
  const float* out_w = (const float*)d_in[6];
  const float* rawmw = (const float*)d_in[7];
  const float* wsc   = (const float*)d_in[8];
  const float* gamma = (const float*)d_in[9];
  float* out = (float*)d_out;

  constexpr size_t NX    = (size_t)MROWS * EE;
  constexpr size_t NQW   = (size_t)EE * EE;
  constexpr size_t NKW   = (size_t)NKV * EE;
  constexpr size_t NKACT = (size_t)MROWS * NKV;

  unsigned char* w = (unsigned char*)d_ws;
  size_t o = 0;
  unsigned short* xb   = (unsigned short*)(w + o); o += NX * 2;
  unsigned short* qwb  = (unsigned short*)(w + o); o += NQW * 2;
  unsigned short* kwb  = (unsigned short*)(w + o); o += NKW * 2;
  unsigned short* vwb  = (unsigned short*)(w + o); o += NKW * 2;
  unsigned short* owb  = (unsigned short*)(w + o); o += NQW * 2;
  unsigned short* qb   = (unsigned short*)(w + o); o += NX * 2;
  unsigned short* kbuf = (unsigned short*)(w + o); o += NKACT * 2;
  unsigned short* vbuf = (unsigned short*)(w + o); o += NKACT * 2;
  unsigned short* nrm  = (unsigned short*)(w + o); o += NX * 2;
  float*          scr  = (float*)(w + o);          o += NX * 4;   // reused f32 scratch

  const dim3 cb(256);
  f32_to_bf16_k<<<4096, cb, 0, stream>>>(x, xb, NX);
  f32_to_bf16_k<<<4096, cb, 0, stream>>>(q_w, qwb, NQW);
  f32_to_bf16_k<<<2048, cb, 0, stream>>>(k_w, kwb, NKW);
  f32_to_bf16_k<<<2048, cb, 0, stream>>>(v_w, vwb, NKW);
  f32_to_bf16_k<<<4096, cb, 0, stream>>>(out_w, owb, NQW);

  // Q projection -> scr (f32) -> RoPE -> qb (bf16)
  gemm_bf16_wmma<<<dim3(EE / BNT, MROWS / BMT), 256, 0, stream>>>(xb, qwb, scr, EE, EE);
  rope_to_bf16<<<(unsigned)((NX / 2 + 255) / 256), 256, 0, stream>>>(
      scr, cosb, sinb, qb, MM * HH, NX / 2);

  // K projection -> RoPE -> kbuf
  gemm_bf16_wmma<<<dim3(NKV / BNT, MROWS / BMT), 256, 0, stream>>>(xb, kwb, scr, NKV, EE);
  rope_to_bf16<<<(unsigned)((NKACT / 2 + 255) / 256), 256, 0, stream>>>(
      scr, cosb, sinb, kbuf, MM * KVH, NKACT / 2);

  // V projection -> vbuf
  gemm_bf16_wmma<<<dim3(NKV / BNT, MROWS / BMT), 256, 0, stream>>>(xb, vwb, scr, NKV, EE);
  f32_to_bf16_k<<<2048, cb, 0, stream>>>(scr, vbuf, NKACT);

  // fused ensemble flash attention -> combined f32 (scr reused)
  flash_ensemble<<<dim3(TT / 16, BB * HH), 128, 0, stream>>>(
      qb, kbuf, vbuf, rawmw, wsc, scr);

  // RMS norm -> bf16
  rmsnorm_bf16<<<(unsigned)(MROWS * HH), 128, 0, stream>>>(scr, gamma, nrm);

  // output projection straight into d_out (f32)
  gemm_bf16_wmma<<<dim3(EE / BNT, MROWS / BMT), 256, 0, stream>>>(nrm, owb, out, EE, EE);
}
